// GIN_5944234737805
// MI455X (gfx1250) — compile-verified, matrix operand order checked by reading, not run on hardware
//
#include <hip/hip_runtime.h>
#include <hip/hip_fp16.h>

#define D 128
#define L_LAYERS 5

typedef __attribute__((ext_vector_type(16))) _Float16 v16h;
typedef __attribute__((ext_vector_type(8)))  _Float16 v8h;
typedef __attribute__((ext_vector_type(8)))  float    v8f;

__device__ __forceinline__ void atomicAddDev(float* p, float v) {
    (void)__hip_atomic_fetch_add(p, v, __ATOMIC_RELAXED, __HIP_MEMORY_SCOPE_AGENT);
}
__device__ __forceinline__ void atomicAddLds(float* p, float v) {
    (void)__hip_atomic_fetch_add(p, v, __ATOMIC_RELAXED, __HIP_MEMORY_SCOPE_WORKGROUP);
}

// ---------------------------------------------------------------------------
// One-time: convert f32 weights (L,D,D) row-major into f16, pre-swizzled into
// WMMA B-fragment order: Wf[l][kt][nt][lane][i] so each lane's fragment is a
// contiguous 32-byte load. Layout per ISA 16-bit operand table:
//   g = lane>>4, N = nt*16 + (lane&15), v = i>>1, j = i&1
//   K = kt*32 + (v&3)*2 + j + 8*g + 16*(v>=4)
// ---------------------------------------------------------------------------
__global__ void gin_wconv(const float* __restrict__ W, _Float16* __restrict__ Wf, int total) {
    int idx = blockIdx.x * 256 + threadIdx.x;
    if (idx >= total) return;
    int i    = idx & 15;
    int lane = (idx >> 4) & 31;
    int nt   = (idx >> 9) & 7;
    int kt   = (idx >> 12) & 3;
    int l    = idx >> 14;
    int g = lane >> 4;
    int n = nt * 16 + (lane & 15);
    int v = i >> 1, j = i & 1;
    int k = kt * 32 + (v & 3) * 2 + j + 8 * g + ((v >= 4) ? 16 : 0);
    Wf[idx] = (_Float16)W[(size_t)l * D * D + (size_t)k * D + n];
}

// Zero agg buffer (as float4) + BN stat accumulators (6*D floats).
__global__ void gin_zero(float4* __restrict__ agg4, int n4, float* __restrict__ stats) {
    int idx = blockIdx.x * 256 + threadIdx.x;
    if (idx < n4) agg4[idx] = make_float4(0.f, 0.f, 0.f, 0.f);
    if (blockIdx.x == 0) {
        for (int s = threadIdx.x; s < 6 * D; s += 256) stats[s] = 0.f;
    }
}

// Edge scatter: one wave per edge, each lane handles 4 consecutive features.
__global__ __launch_bounds__(256) void gin_scatter(const float* __restrict__ h,
                                                   const int* __restrict__ src,
                                                   const int* __restrict__ dst,
                                                   float* __restrict__ agg, int E) {
    int e = blockIdx.x * 8 + (threadIdx.x >> 5);
    if (e >= E) return;
    int lane = threadIdx.x & 31;
    int s = src[e], d = dst[e];
    float4 v = ((const float4*)(h + (size_t)s * D))[lane];
    float* p = agg + (size_t)d * D + lane * 4;
    atomicAddDev(p + 0, v.x);
    atomicAddDev(p + 1, v.y);
    atomicAddDev(p + 2, v.z);
    atomicAddDev(p + 3, v.w);
}

// ---------------------------------------------------------------------------
// Fused GEMM: out = f(A) @ W + bias, plus per-column sum / sumsq for BN.
//   mode 0: f(A) = A0 + A1           (rst = h + agg)
//   mode 1: f(A) = relu(A0*scale+shift)  (BN1+ReLU folded into A-load)
// Block = 256 threads (8 waves) computes a 128x128 output tile.
// A tile converted to f16 in LDS (row pitch 136 halves = 272B, bank-friendly).
// Wave w does rows [16w,16w+16): 4 K-steps x 8 N-tiles of wmma_f32_16x16x32_f16.
// ---------------------------------------------------------------------------
__global__ __launch_bounds__(256) void gin_gemm(
    const float* __restrict__ A0, const float* __restrict__ A1,
    const float* __restrict__ scale, const float* __restrict__ shift,
    const _Float16* __restrict__ Wf, const float* __restrict__ bias,
    float* __restrict__ out, float* __restrict__ ssum, float* __restrict__ ssq,
    int N, int mode)
{
    __shared__ _Float16 As[128][136];
    __shared__ float cs[2][D];
    int tid = threadIdx.x;
    int rowBase = blockIdx.x * 128;
    if (tid < D) { cs[0][tid] = 0.f; cs[1][tid] = 0.f; }

    {   // build f16 A tile: 2 threads per row, 64 columns each
        int r  = tid >> 1;
        int cb = (tid & 1) * 64;
        int grow = rowBase + r;
        if (grow < N) {
            const float4* p0 = (const float4*)(A0 + (size_t)grow * D + cb);
            if (mode == 0) {
                const float4* p1 = (const float4*)(A1 + (size_t)grow * D + cb);
                #pragma unroll
                for (int i = 0; i < 16; ++i) {
                    float4 a = p0[i], b = p1[i];
                    int c = cb + i * 4;
                    As[r][c + 0] = (_Float16)(a.x + b.x);
                    As[r][c + 1] = (_Float16)(a.y + b.y);
                    As[r][c + 2] = (_Float16)(a.z + b.z);
                    As[r][c + 3] = (_Float16)(a.w + b.w);
                }
            } else {
                const float4* sc4 = (const float4*)(scale + cb);
                const float4* sh4 = (const float4*)(shift + cb);
                #pragma unroll
                for (int i = 0; i < 16; ++i) {
                    float4 a = p0[i]; float4 s = sc4[i]; float4 t = sh4[i];
                    int c = cb + i * 4;
                    As[r][c + 0] = (_Float16)fmaxf(a.x * s.x + t.x, 0.f);
                    As[r][c + 1] = (_Float16)fmaxf(a.y * s.y + t.y, 0.f);
                    As[r][c + 2] = (_Float16)fmaxf(a.z * s.z + t.z, 0.f);
                    As[r][c + 3] = (_Float16)fmaxf(a.w * s.w + t.w, 0.f);
                }
            }
        } else {
            #pragma unroll
            for (int i = 0; i < 16; ++i) {
                int c = cb + i * 4;
                As[r][c + 0] = (_Float16)0.f; As[r][c + 1] = (_Float16)0.f;
                As[r][c + 2] = (_Float16)0.f; As[r][c + 3] = (_Float16)0.f;
            }
        }
    }
    __syncthreads();

    int wave = tid >> 5, lane = tid & 31;
    int g = lane >> 4, ln = lane & 15;
    int arow = wave * 16 + ln;   // A-frag: lane holds row M = lane&15 of wave tile

    v8f acc[8];
    v8f vzero = {};
    #pragma unroll
    for (int nt = 0; nt < 8; ++nt) acc[nt] = vzero;

    #pragma unroll
    for (int kt = 0; kt < 4; ++kt) {
        int kb = kt * 32;
        // lane group 0: K = kb+0..7 and kb+16..23 ; group 1: +8
        v8h alo = *(const v8h*)&As[arow][kb + 8 * g];
        v8h ahi = *(const v8h*)&As[arow][kb + 16 + 8 * g];
        v16h af;
        #pragma unroll
        for (int i = 0; i < 8; ++i) { af[i] = alo[i]; af[i + 8] = ahi[i]; }
        #pragma unroll
        for (int nt = 0; nt < 8; ++nt) {
            const _Float16* bp = Wf + ((size_t)(kt * 8 + nt) * 32 + lane) * 16;
            v8h blo = *(const v8h*)bp;
            v8h bhi = *(const v8h*)(bp + 8);
            v16h bf;
            #pragma unroll
            for (int i = 0; i < 8; ++i) { bf[i] = blo[i]; bf[i + 8] = bhi[i]; }
            acc[nt] = __builtin_amdgcn_wmma_f32_16x16x32_f16(
                false, af, false, bf, (short)0, acc[nt], false, false);
        }
    }

    // Epilogue: +bias, store f32, per-column sum/sumsq (C/D layout: M = r + 8g)
    #pragma unroll
    for (int nt = 0; nt < 8; ++nt) {
        int col = nt * 16 + ln;
        float b = bias[col];
        float s = 0.f, q = 0.f;
        int gr0 = rowBase + wave * 16 + 8 * g;
        #pragma unroll
        for (int r = 0; r < 8; ++r) {
            int grow = gr0 + r;
            float v = acc[nt][r] + b;
            if (grow < N) {
                out[(size_t)grow * D + col] = v;
                s += v; q += v * v;
            }
        }
        atomicAddLds(&cs[0][col], s);
        atomicAddLds(&cs[1][col], q);
    }
    __syncthreads();
    if (tid < D) {
        atomicAddDev(ssum + tid, cs[0][tid]);
        atomicAddDev(ssq  + tid, cs[1][tid]);
    }
}

// Fold BN (training-mode, biased var, eps=1e-5) into per-column scale/shift.
__global__ void gin_finalize(const float* __restrict__ ssum, const float* __restrict__ ssq,
                             const float* __restrict__ gamma, const float* __restrict__ beta,
                             float* __restrict__ scale, float* __restrict__ shift, float invN)
{
    int j = threadIdx.x;
    float m   = ssum[j] * invN;
    float var = ssq[j] * invN - m * m;
    float rs  = rsqrtf(var + 1e-5f);
    float sc  = gamma[j] * rs;
    scale[j] = sc;
    shift[j] = beta[j] - m * sc;
}

// y = relu(x*scale+shift) with per-column sum/sumsq for the next BN.
__global__ __launch_bounds__(256) void gin_bnrelu_stats(
    const float* __restrict__ x, const float* __restrict__ scale, const float* __restrict__ shift,
    float* __restrict__ y, float* __restrict__ ssum, float* __restrict__ ssq, int N)
{
    __shared__ float cs[2][D];
    int tid = threadIdx.x;
    int col = tid & (D - 1);
    int rsub = tid >> 7;
    if (tid < D) { cs[0][tid] = 0.f; cs[1][tid] = 0.f; }
    __syncthreads();
    float sc = scale[col], sh = shift[col];
    int rowBase = blockIdx.x * 64;
    float s = 0.f, q = 0.f;
    for (int i = 0; i < 32; ++i) {
        int row = rowBase + rsub + 2 * i;
        if (row < N) {
            size_t idx = (size_t)row * D + col;
            float v = fmaxf(x[idx] * sc + sh, 0.f);
            y[idx] = v;
            s += v; q += v * v;
        }
    }
    atomicAddLds(&cs[0][col], s);
    atomicAddLds(&cs[1][col], q);
    __syncthreads();
    if (tid < D) {
        atomicAddDev(ssum + tid, cs[0][tid]);
        atomicAddDev(ssq  + tid, cs[1][tid]);
    }
}

// Final elementwise BN+ReLU (vectorized float4), no stats needed.
__global__ void gin_bnrelu(const float4* __restrict__ y4, const float* __restrict__ scale,
                           const float* __restrict__ shift, float4* __restrict__ out4, int n4)
{
    int idx = blockIdx.x * 256 + threadIdx.x;
    if (idx >= n4) return;
    int c4 = (idx & 31) * 4;
    float4 v = y4[idx];
    float4 s = *(const float4*)(scale + c4);
    float4 t = *(const float4*)(shift + c4);
    float4 r;
    r.x = fmaxf(v.x * s.x + t.x, 0.f);
    r.y = fmaxf(v.y * s.y + t.y, 0.f);
    r.z = fmaxf(v.z * s.z + t.z, 0.f);
    r.w = fmaxf(v.w * s.w + t.w, 0.f);
    out4[idx] = r;
}

extern "C" void kernel_launch(void* const* d_in, const int* in_sizes, int n_in,
                              void* d_out, int out_size, void* d_ws, size_t ws_size,
                              hipStream_t stream) {
    (void)n_in; (void)out_size; (void)ws_size;
    const float* h0   = (const float*)d_in[0];
    const int*   src  = (const int*)d_in[1];
    const int*   dst  = (const int*)d_in[2];
    const float* W1   = (const float*)d_in[3];
    const float* b1   = (const float*)d_in[4];
    const float* W2   = (const float*)d_in[5];
    const float* b2   = (const float*)d_in[6];
    const float* bn1g = (const float*)d_in[7];
    const float* bn1b = (const float*)d_in[8];
    const float* bn2g = (const float*)d_in[9];
    const float* bn2b = (const float*)d_in[10];
    const float* bn3g = (const float*)d_in[11];
    const float* bn3b = (const float*)d_in[12];

    int N = in_sizes[0] / D;
    int E = in_sizes[1];
    size_t nd = (size_t)N * D;

    // Workspace layout
    float* ws     = (float*)d_ws;
    float* Hbuf   = ws;                  // N*D  current h (layers >= 1)
    float* B0     = Hbuf + nd;           // N*D  agg, then x2
    float* B1     = B0 + nd;             // N*D  x1, then y2
    float* stats  = B1 + nd;             // 6*D  sum1,sq1,sum2,sq2,sum3,sq3
    float* scales = stats + 6 * D;       // 6*D  sc1,sh1,sc2,sh2,sc3,sh3
    _Float16* W1h = (_Float16*)(scales + 6 * D);       // L*D*D f16 (frag order)
    _Float16* W2h = W1h + (size_t)L_LAYERS * D * D;

    int wtotal = L_LAYERS * D * D;
    gin_wconv<<<(wtotal + 255) / 256, 256, 0, stream>>>(W1, W1h, wtotal);
    gin_wconv<<<(wtotal + 255) / 256, 256, 0, stream>>>(W2, W2h, wtotal);

    int n4 = (int)(nd / 4);
    int gemmBlocks = (N + 127) / 128;
    float invN = 1.0f / (float)N;

    const float* hin = h0;
    for (int l = 0; l < L_LAYERS; ++l) {
        gin_zero<<<(n4 + 255) / 256, 256, 0, stream>>>((float4*)B0, n4, stats);
        gin_scatter<<<(E + 7) / 8, 256, 0, stream>>>(hin, src, dst, B0, E);
        // x1 = (h+agg) @ W1 + b1, stats1
        gin_gemm<<<gemmBlocks, 256, 0, stream>>>(hin, B0, nullptr, nullptr,
            W1h + (size_t)l * D * D, b1 + l * D, B1, stats + 0 * D, stats + 1 * D, N, 0);
        gin_finalize<<<1, D, 0, stream>>>(stats + 0 * D, stats + 1 * D,
            bn1g + l * D, bn1b + l * D, scales + 0 * D, scales + 1 * D, invN);
        // x2 = relu(bn1(x1)) @ W2 + b2, stats2
        gin_gemm<<<gemmBlocks, 256, 0, stream>>>(B1, nullptr, scales + 0 * D, scales + 1 * D,
            W2h + (size_t)l * D * D, b2 + l * D, B0, stats + 2 * D, stats + 3 * D, N, 1);
        gin_finalize<<<1, D, 0, stream>>>(stats + 2 * D, stats + 3 * D,
            bn2g + l * D, bn2b + l * D, scales + 2 * D, scales + 3 * D, invN);
        // y2 = relu(bn2(x2)), stats3
        gin_bnrelu_stats<<<(N + 63) / 64, 256, 0, stream>>>(B0, scales + 2 * D, scales + 3 * D,
            B1, stats + 4 * D, stats + 5 * D, N);
        gin_finalize<<<1, D, 0, stream>>>(stats + 4 * D, stats + 5 * D,
            bn3g + l * D, bn3b + l * D, scales + 4 * D, scales + 5 * D, invN);
        // h = relu(bn3(y2))
        float* hout = (l == L_LAYERS - 1) ? (float*)d_out : Hbuf;
        gin_bnrelu<<<(n4 + 255) / 256, 256, 0, stream>>>((const float4*)B1,
            scales + 4 * D, scales + 5 * D, (float4*)hout, n4);
        hin = Hbuf;
    }
}
